// SharedCodebookNWay_56590489092794
// MI455X (gfx1250) — compile-verified, matrix operand order checked by reading, not run on hardware
//
#include <hip/hip_runtime.h>
#include <hip/hip_bf16.h>

// ---------------- problem sizes (fixed by reference) ----------------
#define NROWS 32768
#define DIN   1024
#define DMID  512
#define KCB   1024
#define DECAY_F 0.99f
#define ONE_MINUS_DECAY_F 0.01f
#define EPS_VQ_F 1e-5f
#define EPS_LN_F 1e-5f

typedef __attribute__((ext_vector_type(16))) __bf16 v16bf;
typedef __attribute__((ext_vector_type(8)))  __bf16 v8bf;
typedef __attribute__((ext_vector_type(2)))  __bf16 v2bf;
typedef __attribute__((ext_vector_type(8)))  float  v8f;
typedef __attribute__((ext_vector_type(4)))  unsigned int v4u;
typedef __attribute__((ext_vector_type(8)))  int v8i;
typedef __attribute__((ext_vector_type(4)))  int v4i;

// Fragment = two contiguous 8x bf16 runs: K = half*8 + [0,8) and K = 16 + half*8 + [0,8)
// (ISA 7.12.2 16-bit A/B layout). base must point at K-contiguous row, 16B aligned.
#define LOAD_FRAG(dst, base, k0, half8)                        \
  {                                                            \
    union { v16bf v; v8bf h[2]; } _f;                          \
    _f.h[0] = *(const v8bf*)((base) + (k0) + (half8));         \
    _f.h[1] = *(const v8bf*)((base) + (k0) + 16 + (half8));    \
    (dst) = _f.v;                                              \
  }

// ---------------- small utility kernels ----------------
__global__ void k_f32_to_bf16(const float* __restrict__ s, __bf16* __restrict__ d, int n) {
  for (int i = blockIdx.x * blockDim.x + threadIdx.x; i < n; i += gridDim.x * blockDim.x)
    d[i] = (__bf16)s[i];
}

// dst[n][k] = (bf16) src[k][n]   (src is Kdim x Ndim, row-major)
__global__ void k_transpose_bf16(const float* __restrict__ src, __bf16* __restrict__ dst,
                                 int Kdim, int Ndim) {
  int total = Kdim * Ndim;
  for (int i = blockIdx.x * blockDim.x + threadIdx.x; i < total; i += gridDim.x * blockDim.x) {
    int n = i / Kdim, k = i - n * Kdim;
    dst[i] = (__bf16)src[(size_t)k * Ndim + n];
  }
}

__global__ void k_zero_f32(float* __restrict__ p, int n) {
  for (int i = blockIdx.x * blockDim.x + threadIdx.x; i < n; i += gridDim.x * blockDim.x)
    p[i] = 0.0f;
}

__global__ void k_enorm(const float* __restrict__ emb, float* __restrict__ enorm) {
  int k = blockIdx.x * blockDim.x + threadIdx.x;
  if (k < KCB) {
    float s = 0.f;
    for (int d = 0; d < DMID; ++d) {
      float v = emb[(size_t)k * DMID + d];
      s += v * v;
    }
    enorm[k] = s;
  }
}

// ---------------- encoder: GEMM(x, enc_w) + bias + LayerNorm -> z_e ----------------
// encw_t is n-major: [DMID][DIN] bf16 (K contiguous)
__global__ __launch_bounds__(256) void k_encoder_ln(
    const float* __restrict__ x, const __bf16* __restrict__ encw_t,
    const float* __restrict__ enc_b, const float* __restrict__ ln_g,
    const float* __restrict__ ln_b, float* __restrict__ z_e) {
  __shared__ __bf16 xp[16][DIN];   // 32 KB: whole A panel, staged once
  __shared__ float  h[16][DMID];   // 32 KB
  const int tid  = threadIdx.x;
  const int lane = tid & 31, wave = tid >> 5;
  const int half8 = (lane >> 4) * 8, lm = lane & 15;
  const int m0 = blockIdx.x * 16;

  // stage panel: f32 -> bf16, one float2 per thread per iter
  for (int pe = tid; pe < 16 * DIN / 2; pe += 256) {
    const int r = pe >> 9, c2 = (pe & 511) * 2;
    const float2 v = *(const float2*)(x + (size_t)(m0 + r) * DIN + c2);
    v2bf p; p[0] = (__bf16)v.x; p[1] = (__bf16)v.y;
    *(v2bf*)&xp[r][c2] = p;
  }
  __syncthreads();

  v8f acc[4] = {};
  for (int k0 = 0; k0 < DIN; k0 += 32) {
    v16bf a;
    LOAD_FRAG(a, &xp[lm][0], k0, half8);
#pragma unroll
    for (int t = 0; t < 4; ++t) {
      const int n = 64 * wave + 16 * t + lm;
      const __bf16* bp = encw_t + (size_t)n * DIN;
      __builtin_prefetch(bp + k0 + 32, 0, 1);
      v16bf b;
      LOAD_FRAG(b, bp, k0, half8);
      acc[t] = __builtin_amdgcn_wmma_f32_16x16x32_bf16(false, a, false, b, (short)0,
                                                       acc[t], false, false);
    }
  }
#pragma unroll
  for (int t = 0; t < 4; ++t)
#pragma unroll
    for (int r = 0; r < 8; ++r) {
      const int m = r + half8;               // half8 == 8*half
      const int col = 64 * wave + 16 * t + lm;
      h[m][col] = acc[t][r] + enc_b[col];
    }
  __syncthreads();
  // LayerNorm: 2 rows per wave, full-wave (wave32) shuffle reduction
  for (int rr = 0; rr < 2; ++rr) {
    const int row = wave * 2 + rr;
    float s = 0.f, s2 = 0.f;
#pragma unroll
    for (int j = 0; j < 16; ++j) {
      float v = h[row][lane + 32 * j];
      s += v; s2 += v * v;
    }
#pragma unroll
    for (int off = 16; off > 0; off >>= 1) {
      s  += __shfl_xor(s,  off, 32);
      s2 += __shfl_xor(s2, off, 32);
    }
    const float mu  = s * (1.0f / DMID);
    const float var = s2 * (1.0f / DMID) - mu * mu;
    const float rs  = rsqrtf(var + EPS_LN_F);
#pragma unroll
    for (int j = 0; j < 16; ++j) {
      const int col = lane + 32 * j;
      z_e[(size_t)(m0 + row) * DMID + col] = (h[row][col] - mu) * rs * ln_g[col] + ln_b[col];
    }
  }
}

// ---------------- distance GEMM + per-row argmin ----------------
// embb: [KCB][DMID] bf16 (already K-contiguous per codebook row)
__global__ __launch_bounds__(256) void k_dist_argmin(
    const float* __restrict__ z_e, const __bf16* __restrict__ embb,
    const float* __restrict__ enorm, int* __restrict__ idx_ws,
    int* __restrict__ idx_out) {
  __shared__ __bf16 zp[16][DMID];   // 16 KB panel
  __shared__ float cand_v[16][8];
  __shared__ int   cand_i[16][8];
  const int tid  = threadIdx.x;
  const int lane = tid & 31, wave = tid >> 5;
  const int half8 = (lane >> 4) * 8, lm = lane & 15;
  const int m0 = blockIdx.x * 16;

  for (int pe = tid; pe < 16 * DMID / 2; pe += 256) {
    const int r = pe >> 8, c2 = (pe & 255) * 2;
    const float2 v = *(const float2*)(z_e + (size_t)(m0 + r) * DMID + c2);
    v2bf p; p[0] = (__bf16)v.x; p[1] = (__bf16)v.y;
    *(v2bf*)&zp[r][c2] = p;
  }
  __syncthreads();

  v8f acc[8] = {};
  for (int k0 = 0; k0 < DMID; k0 += 32) {
    v16bf a;
    LOAD_FRAG(a, &zp[lm][0], k0, half8);
#pragma unroll
    for (int t = 0; t < 8; ++t) {
      const int n = 128 * wave + 16 * t + lm;   // codebook column; B[k][n] = emb[n][k]
      const __bf16* bp = embb + (size_t)n * DMID;
      __builtin_prefetch(bp + k0 + 32, 0, 1);
      v16bf b;
      LOAD_FRAG(b, bp, k0, half8);
      acc[t] = __builtin_amdgcn_wmma_f32_16x16x32_bf16(false, a, false, b, (short)0,
                                                       acc[t], false, false);
    }
  }
  // per-lane per-row argmin over this wave's 128 columns (||z||^2 row-constant: dropped)
  float mv[8]; int mi[8];
#pragma unroll
  for (int r = 0; r < 8; ++r) { mv[r] = 3.4e38f; mi[r] = 0; }
#pragma unroll
  for (int t = 0; t < 8; ++t) {
    const int n = 128 * wave + 16 * t + lm;
    const float en = enorm[n];
#pragma unroll
    for (int r = 0; r < 8; ++r) {
      const float val = en - 2.0f * acc[t][r];
      if (val < mv[r]) { mv[r] = val; mi[r] = n; }
    }
  }
#pragma unroll
  for (int off = 1; off < 16; off <<= 1) {
#pragma unroll
    for (int r = 0; r < 8; ++r) {
      float ov = __shfl_xor(mv[r], off, 32);
      int   oi = __shfl_xor(mi[r], off, 32);
      if (ov < mv[r]) { mv[r] = ov; mi[r] = oi; }
    }
  }
  if (lm == 0) {
#pragma unroll
    for (int r = 0; r < 8; ++r) {
      cand_v[r + half8][wave] = mv[r];
      cand_i[r + half8][wave] = mi[r];
    }
  }
  __syncthreads();
  if (tid < 16) {
    float best = 3.4e38f; int bi = 0;
#pragma unroll
    for (int w = 0; w < 8; ++w)
      if (cand_v[tid][w] < best) { best = cand_v[tid][w]; bi = cand_i[tid][w]; }
    idx_ws[m0 + tid]  = bi;
    idx_out[m0 + tid] = bi;
  }
}

// ---------------- scatter: counts, embed_sum, z_q_st gather, commit-loss partials ------
__global__ __launch_bounds__(256) void k_scatter(
    const int* __restrict__ idx_ws, const float* __restrict__ z_e,
    const float* __restrict__ emb, float* __restrict__ counts,
    float* __restrict__ embsum, float* __restrict__ zq_out,
    float* __restrict__ commit_acc) {
  __shared__ float red[256];
  float local = 0.f;
  const int total = NROWS * DMID;
  for (int i = blockIdx.x * blockDim.x + threadIdx.x; i < total;
       i += gridDim.x * blockDim.x) {
    const int n = i >> 9;
    const int d = i & 511;
    const int id = idx_ws[n];
    const float ze = z_e[i];
    const float zq = emb[(size_t)id * DMID + d];
    zq_out[i] = zq;                                    // z_q_st == z_q in value
    atomicAdd(&embsum[(size_t)id * DMID + d], ze);
    const float df = ze - zq;
    local += df * df;
    if (d == 0) atomicAdd(&counts[id], 1.0f);
  }
  red[threadIdx.x] = local;
  __syncthreads();
  for (int s = 128; s > 0; s >>= 1) {
    if (threadIdx.x < s) red[threadIdx.x] += red[threadIdx.x + s];
    __syncthreads();
  }
  if (threadIdx.x == 0) atomicAdd(commit_acc, red[0]);
}

__global__ void k_commit_finalize(const float* __restrict__ acc, float* __restrict__ out) {
  out[0] = acc[0] * (1.0f / ((float)NROWS * (float)DMID));
}

// ---------------- EMA updates ----------------
__global__ void k_ema1(const float* __restrict__ cs_in, const float* __restrict__ counts,
                       float* __restrict__ ncs_out, float* __restrict__ nsum) {
  int k = blockIdx.x * blockDim.x + threadIdx.x;
  if (k < KCB) {
    float v = DECAY_F * cs_in[k] + ONE_MINUS_DECAY_F * counts[k];
    ncs_out[k] = v;
    atomicAdd(nsum, v);
  }
}

__global__ void k_ema2(const float* __restrict__ ema_in, const float* __restrict__ embsum,
                       const float* __restrict__ ncs, const float* __restrict__ nsum,
                       float* __restrict__ nema_out, float* __restrict__ nemb_out) {
  int i = blockIdx.x * blockDim.x + threadIdx.x;
  if (i < KCB * DMID) {
    const int k = i >> 9;
    const float nema = DECAY_F * ema_in[i] + ONE_MINUS_DECAY_F * embsum[i];
    nema_out[i] = nema;
    const float n = nsum[0];
    const float csk = (ncs[k] + EPS_VQ_F) / (n + (float)KCB * EPS_VQ_F) * n;
    nemb_out[i] = nema / csk;
  }
}

// ---------------- decoder: x_recon = gather(emb, idx) @ dec_w + dec_b ----------------
// A panel gathered by the Tensor Data Mover in gather mode (16x 16-bit row indices).
// decw_t is n-major: [DIN][DMID] bf16.
__global__ __launch_bounds__(256) void k_decoder(
    const int* __restrict__ idx_ws, const __bf16* __restrict__ embb,
    const __bf16* __restrict__ decw_t, const float* __restrict__ dec_b,
    float* __restrict__ xrec) {
  __shared__ __bf16 qp[16][DMID];   // 16 KB gathered A panel
  const int tid  = threadIdx.x;
  const int lane = tid & 31, wave = tid >> 5;
  const int half8 = (lane >> 4) * 8, lm = lane & 15;
  const int m0 = blockIdx.x * 16;

  if (wave == 0) {
    // Build Tensor DMA descriptor (D#), gather mode, per ISA 8.3-8.7.
    const int* ip = idx_ws + m0;   // block-uniform scalar loads
    unsigned long long ga = (unsigned long long)(const void*)embb;
    unsigned lds = (unsigned)(size_t)(&qp[0][0]);

    v4u g0;
    g0[0] = 0x80000001u;                              // count=1, gather_mode=1, 16-bit idx
    g0[1] = lds;                                      // lds_addr
    g0[2] = (unsigned)(ga & 0xFFFFFFFFu);             // global_addr[31:0]
    g0[3] = (unsigned)((ga >> 32) & 0x01FFFFFFu) | (2u << 30);  // addr[56:32] | type=2

    v8i g1;
    g1[0] = 0x00010000;            // workgroup_mask=0, data_size=1 (2 bytes)
    g1[1] = (int)(DMID << 16);     // tensor_dim0[15:0]=512 in bits[31:16]
    g1[2] = (int)(KCB << 16);      // tensor_dim0 hi=0 | tensor_dim1[15:0]=1024
    g1[3] = (int)(DMID << 16);     // tensor_dim1 hi=0 | tile_dim0=512
    g1[4] = 16;                    // tile_dim1 = 16 gather indices; tile_dim2=0
    g1[5] = DMID;                  // tensor_dim0_stride = 512 (low 32)
    g1[6] = 0;
    g1[7] = 0;

    v4i g2, g3;
#pragma unroll
    for (int j = 0; j < 4; ++j)
      g2[j] = (ip[2 * j] & 0xFFFF) | (ip[2 * j + 1] << 16);
#pragma unroll
    for (int j = 0; j < 4; ++j)
      g3[j] = (ip[8 + 2 * j] & 0xFFFF) | (ip[8 + 2 * j + 1] << 16);

    v8i g4 = {0, 0, 0, 0, 0, 0, 0, 0};   // 6-arg toolchain variant: extra group, unused

    __builtin_amdgcn_tensor_load_to_lds(g0, g1, g2, g3, g4, 0);
    __builtin_amdgcn_s_wait_tensorcnt(0);
  }
  __syncthreads();

  v8f acc[8] = {};
  for (int k0 = 0; k0 < DMID; k0 += 32) {
    v16bf a;
    LOAD_FRAG(a, &qp[lm][0], k0, half8);
#pragma unroll
    for (int t = 0; t < 8; ++t) {
      const int n = 128 * wave + 16 * t + lm;
      const __bf16* bp = decw_t + (size_t)n * DMID;
      __builtin_prefetch(bp + k0 + 32, 0, 1);
      v16bf b;
      LOAD_FRAG(b, bp, k0, half8);
      acc[t] = __builtin_amdgcn_wmma_f32_16x16x32_bf16(false, a, false, b, (short)0,
                                                       acc[t], false, false);
    }
  }
#pragma unroll
  for (int t = 0; t < 8; ++t)
#pragma unroll
    for (int r = 0; r < 8; ++r) {
      const int m = r + half8;
      const int col = 128 * wave + 16 * t + lm;
      xrec[(size_t)(m0 + m) * DIN + col] = acc[t][r] + dec_b[col];
    }
}

// ---------------- host launcher ----------------
extern "C" void kernel_launch(void* const* d_in, const int* in_sizes, int n_in,
                              void* d_out, int out_size, void* d_ws, size_t ws_size,
                              hipStream_t stream) {
  (void)in_sizes; (void)n_in; (void)out_size; (void)ws_size;
  const float* x      = (const float*)d_in[0];
  const float* enc_w  = (const float*)d_in[1];
  const float* enc_b  = (const float*)d_in[2];
  const float* ln_g   = (const float*)d_in[3];
  const float* ln_b   = (const float*)d_in[4];
  const float* dec_w  = (const float*)d_in[5];
  const float* dec_b  = (const float*)d_in[6];
  const float* emb    = (const float*)d_in[7];
  const float* cs_in  = (const float*)d_in[8];
  const float* ema_in = (const float*)d_in[9];

  float* outf = (float*)d_out;
  const size_t OFF_XREC = 0;
  const size_t OFF_LOSS = OFF_XREC + (size_t)NROWS * DIN;
  const size_t OFF_IDX  = OFF_LOSS + 1;
  const size_t OFF_ZE   = OFF_IDX + NROWS;
  const size_t OFF_ZQ   = OFF_ZE + (size_t)NROWS * DMID;
  const size_t OFF_NEMB = OFF_ZQ + (size_t)NROWS * DMID;
  const size_t OFF_NCS  = OFF_NEMB + (size_t)KCB * DMID;
  const size_t OFF_NEMA = OFF_NCS + KCB;

  float* xrec_out = outf + OFF_XREC;
  float* loss_out = outf + OFF_LOSS;
  int*   idx_out  = (int*)(outf + OFF_IDX);
  float* ze_out   = outf + OFF_ZE;
  float* zq_out   = outf + OFF_ZQ;
  float* nemb_out = outf + OFF_NEMB;
  float* ncs_out  = outf + OFF_NCS;
  float* nema_out = outf + OFF_NEMA;

  char* w = (char*)d_ws;
  size_t off = 0;
  auto carve = [&](size_t bytes) { void* p = w + off; off = (off + bytes + 255) & ~(size_t)255; return p; };
  __bf16* encw_t  = (__bf16*)carve((size_t)DMID * DIN * 2);   // [DMID][DIN]
  __bf16* decw_t  = (__bf16*)carve((size_t)DIN * DMID * 2);   // [DIN][DMID]
  __bf16* emb_b   = (__bf16*)carve((size_t)KCB * DMID * 2);   // [KCB][DMID]
  float*  enorm   = (float*)carve((size_t)KCB * 4);
  float*  counts  = (float*)carve((size_t)KCB * 4);
  float*  embsum  = (float*)carve((size_t)KCB * DMID * 4);
  int*    idx_ws  = (int*)carve((size_t)NROWS * 4);
  float*  nsum    = (float*)carve(256);
  float*  commit  = (float*)carve(256);

  // phase 0: conversions/transposes, norms, zeroed accumulators
  k_transpose_bf16<<<1024, 256, 0, stream>>>(enc_w, encw_t, DIN, DMID);   // [DMID][DIN]
  k_transpose_bf16<<<1024, 256, 0, stream>>>(dec_w, decw_t, DMID, DIN);   // [DIN][DMID]
  k_f32_to_bf16<<<1024, 256, 0, stream>>>(emb, emb_b, KCB * DMID);
  k_enorm<<<(KCB + 255) / 256, 256, 0, stream>>>(emb, enorm);
  k_zero_f32<<<512, 256, 0, stream>>>(counts, KCB);
  k_zero_f32<<<2048, 256, 0, stream>>>(embsum, KCB * DMID);
  k_zero_f32<<<1, 1, 0, stream>>>(nsum, 1);
  k_zero_f32<<<1, 1, 0, stream>>>(commit, 1);

  // phase 1: encoder + layernorm
  k_encoder_ln<<<NROWS / 16, 256, 0, stream>>>(x, encw_t, enc_b, ln_g, ln_b, ze_out);
  // phase 2: distances + argmin
  k_dist_argmin<<<NROWS / 16, 256, 0, stream>>>(ze_out, emb_b, enorm, idx_ws, idx_out);
  // phase 3: scatter
  k_scatter<<<8192, 256, 0, stream>>>(idx_ws, ze_out, emb, counts, embsum, zq_out, commit);
  k_commit_finalize<<<1, 1, 0, stream>>>(commit, loss_out);
  // phase 4: EMA updates
  k_ema1<<<(KCB + 255) / 256, 256, 0, stream>>>(cs_in, counts, ncs_out, nsum);
  k_ema2<<<(KCB * DMID + 255) / 256, 256, 0, stream>>>(ema_in, embsum, ncs_out, nsum,
                                                       nema_out, nemb_out);
  // phase 5: decoder (TDM-gathered A panel)
  k_decoder<<<NROWS / 16, 256, 0, stream>>>(idx_ws, emb_b, decw_t, dec_b, xrec_out);
}
